// PatchSelector_687194768111
// MI455X (gfx1250) — matrix operand order repaired and checked by reading, not compiled, not run on hardware
//
#include <hip/hip_runtime.h>
#include <hip/hip_bf16.h>
#include <math.h>

typedef __attribute__((ext_vector_type(16))) _Float16 v16h;
typedef __attribute__((ext_vector_type(8)))  float    v8f;

#define BQ   8
#define NQ   8192
#define DQ   768
#define HQ   12
#define HDQ  64
#define HD2Q 128
#define MQ   (BQ * NQ)          // 65536 rows
#define KSEL (NQ / 2)           // 4096

// ---------------- workspace layout (bytes) ----------------
#define WS_STATS   0            // [M][2] f32                     (524288)
#define WS_W1F     524288       // [1536][768] f16 n-major        (2359296)
#define WS_W2F     2883584      // [12][64][128] f16 n-major      (196608)
#define WS_SCORES  3080192      // [M][12] f32                    (3145728)
#define WS_FUSED   6225920      // [M] f32                        (262144)
#define WS_RANK    6488064      // [M] i32                        (262144)

// ---------------- weight prep: w1 (H,D,2HD) -> [h*128+e][k] f16 ----------------
__global__ void prep_w1_kernel(const float* __restrict__ w1, _Float16* __restrict__ w1f) {
    int tid = blockIdx.x * 256 + threadIdx.x;
    if (tid < HQ * DQ * HD2Q) {
        int h = tid / (DQ * HD2Q);
        int rem = tid % (DQ * HD2Q);
        int k = rem / HD2Q;
        int e = rem % HD2Q;
        w1f[(size_t)(h * HD2Q + e) * DQ + k] = (_Float16)w1[tid];
    }
}

// w2 (H,128,64) -> [h*64+e][k] f16
__global__ void prep_w2_kernel(const float* __restrict__ w2, _Float16* __restrict__ w2f) {
    int tid = blockIdx.x * 256 + threadIdx.x;
    if (tid < HQ * HD2Q * HDQ) {
        int h = tid / (HD2Q * HDQ);
        int k = (tid / HDQ) % HD2Q;
        int e = tid % HDQ;
        w2f[(size_t)(h * HDQ + e) * HD2Q + k] = (_Float16)w2[tid];
    }
}

// ---------------- per-row LayerNorm stats of patches ----------------
__global__ __launch_bounds__(256)
void stats_kernel(const float* __restrict__ patches, float* __restrict__ stats) {
    __shared__ float rs[256], rq[256];
    int row = blockIdx.x;
    int t = threadIdx.x;
    const float* p = patches + (size_t)row * DQ;
    float s = 0.f, q = 0.f;
    for (int i = t; i < DQ; i += 256) { float v = p[i]; s += v; q += v * v; }
    rs[t] = s; rq[t] = q; __syncthreads();
    for (int st = 128; st > 0; st >>= 1) {
        if (t < st) { rs[t] += rs[t + st]; rq[t] += rq[t + st]; }
        __syncthreads();
    }
    if (t == 0) {
        float m = rs[0] * (1.f / DQ);
        float var = rq[0] * (1.f / DQ) - m * m;
        stats[2 * row]     = m;
        stats[2 * row + 1] = rsqrtf(var + 1e-5f);
    }
}

__device__ __forceinline__ float gelu_exact(float x) {
    return 0.5f * x * (1.0f + erff(x * 0.70710678118654752f));
}

// ---------------- fused LN -> GEMM1 -> LN+GELU -> GEMM2 -> LN+GELU -> score ----------------
// grid: (M/64, H), block 256 (8 wave32). Each block: 64 rows x one head.
#define AS_STRIDE  48      // f16 (32 cols + pad), 96B row -> 16B aligned frags
#define AS_BYTES   (64 * AS_STRIDE * 2)   // 6144 per buffer
#define C1_STRIDE  132     // f32
#define H1_STRIDE  136     // f16, 272B row -> 16B aligned frags
#define C2_STRIDE  68      // f32

#define SM_REG1    33792                 // As[2] / C1s / C2s (aliased across barriers)
#define SM_H1      33792                 // H1s offset
#define SM_RED     (33792 + 17408)       // 256*2 f32
#define SM_RST     (33792 + 17408 + 2048)// 64*2 f32
#define SM_TOTAL   (33792 + 17408 + 2048 + 512)

// LN + f32->f16 staging of a 64x32 tile (each thread: 8 contiguous cols of one row)
__device__ __forceinline__ void stage_tile(_Float16* __restrict__ dst,
                                           const float* __restrict__ prow,
                                           const float* __restrict__ lng,
                                           const float* __restrict__ lnb,
                                           int kb, int sr, int sc,
                                           float smean, float srstd) {
    const float4 p0  = *(const float4*)(prow + kb + sc);
    const float4 p1  = *(const float4*)(prow + kb + sc + 4);
    const float4 g0  = *(const float4*)(lng + kb + sc);
    const float4 g1  = *(const float4*)(lng + kb + sc + 4);
    const float4 bb0 = *(const float4*)(lnb + kb + sc);
    const float4 bb1 = *(const float4*)(lnb + kb + sc + 4);
    union { _Float16 hh[8]; uint4 u; } t;
    t.hh[0] = (_Float16)((p0.x - smean) * srstd * g0.x + bb0.x);
    t.hh[1] = (_Float16)((p0.y - smean) * srstd * g0.y + bb0.y);
    t.hh[2] = (_Float16)((p0.z - smean) * srstd * g0.z + bb0.z);
    t.hh[3] = (_Float16)((p0.w - smean) * srstd * g0.w + bb0.w);
    t.hh[4] = (_Float16)((p1.x - smean) * srstd * g1.x + bb1.x);
    t.hh[5] = (_Float16)((p1.y - smean) * srstd * g1.y + bb1.y);
    t.hh[6] = (_Float16)((p1.z - smean) * srstd * g1.z + bb1.z);
    t.hh[7] = (_Float16)((p1.w - smean) * srstd * g1.w + bb1.w);
    *(uint4*)(dst + sr * AS_STRIDE + sc) = t.u;
}

__device__ __forceinline__ v16h load_bfrag(const _Float16* __restrict__ bp) {
    v16h bf;
    ((uint4*)&bf)[0] = *(const uint4*)bp;
    ((uint4*)&bf)[1] = *(const uint4*)(bp + 8);
    return bf;
}

__global__ __launch_bounds__(256)
void mega_kernel(const float* __restrict__ patches,
                 const float* __restrict__ stats,
                 const float* __restrict__ lng,  const float* __restrict__ lnb,
                 const _Float16* __restrict__ w1f,
                 const float* __restrict__ b1,
                 const float* __restrict__ ln1g, const float* __restrict__ ln1b,
                 const _Float16* __restrict__ w2f,
                 const float* __restrict__ b2,
                 const float* __restrict__ ln2g, const float* __restrict__ ln2b,
                 const float* __restrict__ w3,   const float* __restrict__ b3,
                 float* __restrict__ scores) {
    __shared__ __align__(16) char smem[SM_TOTAL];
    _Float16* As0 = (_Float16*)smem;
    _Float16* As1 = (_Float16*)(smem + AS_BYTES);
    float*    C1s = (float*)smem;
    float*    C2s = (float*)smem;
    _Float16* H1s = (_Float16*)(smem + SM_H1);
    float*    red = (float*)(smem + SM_RED);
    float*    rst = (float*)(smem + SM_RST);

    const int tid   = threadIdx.x;
    const int lane  = tid & 31;
    const int wave  = tid >> 5;
    const int wm    = wave & 3;          // row group (4 x 16 rows)
    const int wn    = wave >> 2;         // col group (2 x 64 cols)
    const int lm    = lane & 15;
    const int khalf = lane >> 4;
    const int h     = blockIdx.y;
    const int row0  = blockIdx.x * 64;

    // stage-A mapping: each thread loads 8 contiguous cols of one row
    const int sr = tid >> 2;             // 0..63
    const int sc = (tid & 3) * 8;        // 0,8,16,24
    const int grow = row0 + sr;
    const float smean = stats[2 * grow];
    const float srstd = stats[2 * grow + 1];
    const float* prow = patches + (size_t)grow * DQ;

    const int mrow = wm * 16 + lm;       // local row of this lane's A fragment

    v8f acc[4] = {};

    // prime buffer 0
    stage_tile(As0, prow, lng, lnb, 0, sr, sc, smean, srstd);
    __syncthreads();

    int buf = 0;
    for (int kb = 0; kb < DQ; kb += 32) {
        const _Float16* Acur = buf ? As1 : As0;
        _Float16*       Anxt = buf ? As0 : As1;

        // prefetch patches two K-steps ahead (global_prefetch_b8)
        if (kb + 64 < DQ) __builtin_prefetch(prow + kb + 64 + sc, 0, 1);

        // stage next K tile into the other buffer (overlaps with WMMAs below)
        if (kb + 32 < DQ)
            stage_tile(Anxt, prow, lng, lnb, kb + 32, sr, sc, smean, srstd);

        // ---- batch all 4 B fragments (8x global_load_b128, L2-hot) ----
        v16h bf0, bf1, bf2, bf3;
        {
            const size_t nb = (size_t)(h * HD2Q + wn * 64 + lm) * DQ + kb + khalf * 16;
            bf0 = load_bfrag(w1f + nb);
            bf1 = load_bfrag(w1f + nb + (size_t)16 * DQ);
            bf2 = load_bfrag(w1f + nb + (size_t)32 * DQ);
            bf3 = load_bfrag(w1f + nb + (size_t)48 * DQ);
        }

        // ---- A fragment (two ds_load_b128, contiguous k = 16*khalf + s, matching B) ----
        v16h a;
        ((uint4*)&a)[0] = *(const uint4*)(Acur + mrow * AS_STRIDE + khalf * 16);
        ((uint4*)&a)[1] = *(const uint4*)(Acur + mrow * AS_STRIDE + khalf * 16 + 8);

        acc[0] = __builtin_amdgcn_wmma_f32_16x16x32_f16(false, a, false, bf0, (short)0, acc[0], false, false);
        acc[1] = __builtin_amdgcn_wmma_f32_16x16x32_f16(false, a, false, bf1, (short)0, acc[1], false, false);
        acc[2] = __builtin_amdgcn_wmma_f32_16x16x32_f16(false, a, false, bf2, (short)0, acc[2], false, false);
        acc[3] = __builtin_amdgcn_wmma_f32_16x16x32_f16(false, a, false, bf3, (short)0, acc[3], false, false);

        __syncthreads();     // single barrier per K-step (double-buffered A)
        buf ^= 1;
    }

    // ---- dump GEMM1 accumulators to LDS (C layout: VGPR v -> M=v / v+8) ----
    #pragma unroll
    for (int t = 0; t < 4; t++) {
        #pragma unroll
        for (int v = 0; v < 8; v++) {
            int ml = wm * 16 + v + khalf * 8;
            int nc = wn * 64 + t * 16 + lm;
            C1s[ml * C1_STRIDE + nc] = acc[t][v];
        }
    }
    __syncthreads();

    // ---- LN1 + GELU -> H1s (f16) ----
    {
        const int r = tid >> 2, q = tid & 3, cb = q * 32;
        const float* b1h = b1 + h * HD2Q;
        float vloc[32], s = 0.f, sq = 0.f;
        #pragma unroll
        for (int i = 0; i < 32; i++) {
            float v = C1s[r * C1_STRIDE + cb + i] + b1h[cb + i];
            vloc[i] = v; s += v; sq += v * v;
        }
        red[2 * tid] = s; red[2 * tid + 1] = sq;
        __syncthreads();
        if (q == 0) {
            float ts = 0.f, tq = 0.f;
            #pragma unroll
            for (int j = 0; j < 4; j++) { ts += red[2 * (tid + j)]; tq += red[2 * (tid + j) + 1]; }
            float m = ts * (1.f / HD2Q);
            float var = tq * (1.f / HD2Q) - m * m;
            rst[2 * r] = m; rst[2 * r + 1] = rsqrtf(var + 1e-5f);
        }
        __syncthreads();
        const float m = rst[2 * r], rs = rst[2 * r + 1];
        const float* g  = ln1g + h * HD2Q;
        const float* bb = ln1b + h * HD2Q;
        #pragma unroll
        for (int i = 0; i < 32; i++) {
            int c = cb + i;
            float v = (vloc[i] - m) * rs * g[c] + bb[c];
            H1s[r * H1_STRIDE + c] = (_Float16)gelu_exact(v);
        }
    }
    __syncthreads();

    // ---- GEMM2: 64x64, K=128, A from H1s (LDS), B = w2f (L2-hot) ----
    v8f acc2[2] = {};
    #pragma unroll
    for (int kb2 = 0; kb2 < HD2Q; kb2 += 32) {
        const size_t nb = (size_t)(h * HDQ + wn * 32 + lm) * HD2Q + kb2 + khalf * 16;
        v16h bg0 = load_bfrag(w2f + nb);
        v16h bg1 = load_bfrag(w2f + nb + (size_t)16 * HD2Q);
        v16h a2;
        ((uint4*)&a2)[0] = *(const uint4*)(H1s + mrow * H1_STRIDE + kb2 + khalf * 16);
        ((uint4*)&a2)[1] = *(const uint4*)(H1s + mrow * H1_STRIDE + kb2 + khalf * 16 + 8);
        acc2[0] = __builtin_amdgcn_wmma_f32_16x16x32_f16(false, a2, false, bg0, (short)0, acc2[0], false, false);
        acc2[1] = __builtin_amdgcn_wmma_f32_16x16x32_f16(false, a2, false, bg1, (short)0, acc2[1], false, false);
    }
    __syncthreads();   // all H1s reads done; C2s aliases region1 (C1s dead)

    #pragma unroll
    for (int t = 0; t < 2; t++) {
        #pragma unroll
        for (int v = 0; v < 8; v++) {
            int ml = wm * 16 + v + khalf * 8;
            int nc = wn * 32 + t * 16 + lm;
            C2s[ml * C2_STRIDE + nc] = acc2[t][v];
        }
    }
    __syncthreads();

    // ---- LN2 + GELU + dot(w3) -> scores ----
    {
        const int r = tid >> 2, q = tid & 3, cb = q * 16;
        const float* b2h = b2 + h * HDQ;
        float vloc[16], s = 0.f, sq = 0.f;
        #pragma unroll
        for (int i = 0; i < 16; i++) {
            float v = C2s[r * C2_STRIDE + cb + i] + b2h[cb + i];
            vloc[i] = v; s += v; sq += v * v;
        }
        red[2 * tid] = s; red[2 * tid + 1] = sq;
        __syncthreads();
        if (q == 0) {
            float ts = 0.f, tq = 0.f;
            #pragma unroll
            for (int j = 0; j < 4; j++) { ts += red[2 * (tid + j)]; tq += red[2 * (tid + j) + 1]; }
            float m = ts * (1.f / HDQ);
            float var = tq * (1.f / HDQ) - m * m;
            rst[2 * r] = m; rst[2 * r + 1] = rsqrtf(var + 1e-5f);
        }
        __syncthreads();
        const float m = rst[2 * r], rs = rst[2 * r + 1];
        const float* g   = ln2g + h * HDQ;
        const float* bb  = ln2b + h * HDQ;
        const float* w3h = w3   + h * HDQ;
        float dp = 0.f;
        #pragma unroll
        for (int i = 0; i < 16; i++) {
            int c = cb + i;
            float v = (vloc[i] - m) * rs * g[c] + bb[c];
            dp += gelu_exact(v) * w3h[c];
        }
        __syncthreads();
        red[tid] = dp;
        __syncthreads();
        if (q == 0) {
            float sc = red[4 * r] + red[4 * r + 1] + red[4 * r + 2] + red[4 * r + 3] + b3[h];
            scores[(size_t)(row0 + r) * HQ + h] = sc;
        }
    }
}

// ---------------- fuse: sigmoid(scores @ fw + fb) ----------------
__global__ void fuse_kernel(const float* __restrict__ scores,
                            const float* __restrict__ fw, const float* __restrict__ fb,
                            float* __restrict__ fused) {
    int i = blockIdx.x * 256 + threadIdx.x;
    if (i < MQ) {
        float z = fb[0];
        #pragma unroll
        for (int h = 0; h < HQ; h++) z += scores[(size_t)i * HQ + h] * fw[h];
        fused[i] = 1.0f / (1.0f + expf(-z));
    }
}

// ---------------- per-batch softmax + stable descending rank ----------------
__global__ __launch_bounds__(1024)
void softmax_rank_kernel(const float* __restrict__ fused,
                         float* __restrict__ attn_out, int* __restrict__ rank) {
    __shared__ float vals[NQ];
    __shared__ float red[1024];
    const int b = blockIdx.x, t = threadIdx.x;
    for (int i = t; i < NQ; i += 1024) vals[i] = fused[b * NQ + i];
    __syncthreads();
    float mx = -1e30f;
    for (int i = t; i < NQ; i += 1024) mx = fmaxf(mx, vals[i]);
    red[t] = mx; __syncthreads();
    for (int s = 512; s > 0; s >>= 1) {
        if (t < s) red[t] = fmaxf(red[t], red[t + s]);
        __syncthreads();
    }
    mx = red[0]; __syncthreads();
    float sm = 0.f;
    for (int i = t; i < NQ; i += 1024) sm += expf(vals[i] - mx);
    red[t] = sm; __syncthreads();
    for (int s = 512; s > 0; s >>= 1) {
        if (t < s) red[t] += red[t + s];
        __syncthreads();
    }
    const float inv = 1.0f / red[0];
    for (int i = t; i < NQ; i += 1024)
        attn_out[b * NQ + i] = expf(vals[i] - mx) * inv;
    // rank: #(v_j > v_i) + #(v_j == v_i && j < i)  -> matches top_k stable order
    for (int i = t; i < NQ; i += 1024) {
        float vi = vals[i];
        int cnt = 0;
        for (int j = 0; j < NQ; j++) {
            float vj = vals[j];
            cnt += (vj > vi) | ((vj == vi) & (j < i));
        }
        rank[b * NQ + i] = cnt;
    }
}

// ---------------- gather top-K patch rows (float4 copy) ----------------
__global__ __launch_bounds__(192)
void gather_kernel(const float* __restrict__ patches, const int* __restrict__ rank,
                   float* __restrict__ out) {
    const int b = blockIdx.x >> 13;
    const int i = blockIdx.x & (NQ - 1);
    const int r = rank[b * NQ + i];
    if (r < KSEL) {
        const float4* src = (const float4*)(patches + (size_t)(b * NQ + i) * DQ);
        float4* dst = (float4*)(out + ((size_t)b * KSEL + r) * DQ);
        dst[threadIdx.x] = src[threadIdx.x];
    }
}

extern "C" void kernel_launch(void* const* d_in, const int* in_sizes, int n_in,
                              void* d_out, int out_size, void* d_ws, size_t ws_size,
                              hipStream_t stream) {
    const float* patches = (const float*)d_in[0];
    const float* lng  = (const float*)d_in[1];
    const float* lnb  = (const float*)d_in[2];
    const float* w1   = (const float*)d_in[3];
    const float* b1   = (const float*)d_in[4];
    const float* ln1g = (const float*)d_in[5];
    const float* ln1b = (const float*)d_in[6];
    const float* w2   = (const float*)d_in[7];
    const float* b2   = (const float*)d_in[8];
    const float* ln2g = (const float*)d_in[9];
    const float* ln2b = (const float*)d_in[10];
    const float* w3   = (const float*)d_in[11];
    const float* b3   = (const float*)d_in[12];
    const float* fw   = (const float*)d_in[13];
    const float* fb   = (const float*)d_in[14];

    char* ws = (char*)d_ws;
    float*     stats  = (float*)(ws + WS_STATS);
    _Float16*  w1f    = (_Float16*)(ws + WS_W1F);
    _Float16*  w2f    = (_Float16*)(ws + WS_W2F);
    float*     scores = (float*)(ws + WS_SCORES);
    float*     fused  = (float*)(ws + WS_FUSED);
    int*       rank   = (int*)(ws + WS_RANK);

    float* out      = (float*)d_out;
    float* attn_out = out + (size_t)BQ * KSEL * DQ;

    prep_w1_kernel<<<(HQ * DQ * HD2Q + 255) / 256, 256, 0, stream>>>(w1, w1f);
    prep_w2_kernel<<<(HQ * HD2Q * HDQ + 255) / 256, 256, 0, stream>>>(w2, w2f);
    stats_kernel<<<MQ, 256, 0, stream>>>(patches, stats);

    dim3 grid(MQ / 64, HQ);
    mega_kernel<<<grid, 256, 0, stream>>>(patches, stats, lng, lnb,
                                          w1f, b1, ln1g, ln1b,
                                          w2f, b2, ln2g, ln2b,
                                          w3, b3, scores);

    fuse_kernel<<<(MQ + 255) / 256, 256, 0, stream>>>(scores, fw, fb, fused);
    softmax_rank_kernel<<<BQ, 1024, 0, stream>>>(fused, attn_out, rank);
    gather_kernel<<<BQ * NQ, 192, 0, stream>>>(patches, rank, out);
}
